// ResidualVectorQuantizer_84585085927521
// MI455X (gfx1250) — matrix operand order repaired and checked by reading, not compiled, not run on hardware
//
#include <hip/hip_runtime.h>
#include <hip/hip_bf16.h>

#define B_ROWS   65536
#define E_DIM    256
#define NE_CODES 1024
#define L_STAGES 4
#define BETA_F   0.55f

typedef __attribute__((ext_vector_type(16))) __bf16         v16bf;
typedef __attribute__((ext_vector_type(16))) unsigned short v16us;
typedef __attribute__((ext_vector_type(8)))  float          v8f;

__device__ __forceinline__ unsigned short f2bf(float f) {
  unsigned u = __builtin_bit_cast(unsigned, f);
  unsigned r = u + 0x7FFFu + ((u >> 16) & 1u);   // round-to-nearest-even
  return (unsigned short)(r >> 16);
}

// ---------------------------------------------------------------------------
// Prep: codebooks f32 -> bf16 bits in ws; hn[l][code] = 0.5*||c_bf16||^2
// One wave per codebook row (L*NE = 4096 rows), 4 waves per block.
// ---------------------------------------------------------------------------
__global__ __launch_bounds__(128)
void rvq_prep(const float* __restrict__ cb, unsigned short* __restrict__ cbw,
              float* __restrict__ hn) {
  const int wave = threadIdx.x >> 5;
  const int lane = threadIdx.x & 31;
  const int rowid = blockIdx.x * 4 + wave;              // 0 .. L*NE-1
  const float* src = cb + (size_t)rowid * E_DIM;
  unsigned short* dst = cbw + (size_t)rowid * E_DIM;
  float ss = 0.f;
#pragma unroll
  for (int j = 0; j < 8; ++j) {
    const int e = lane + 32 * j;
    const unsigned short bits = f2bf(src[e]);
    dst[e] = bits;
    const float bv = __builtin_bit_cast(float, (unsigned)bits << 16);
    ss += bv * bv;
  }
#pragma unroll
  for (int m = 16; m >= 1; m >>= 1) ss += __shfl_xor(ss, m, 32);
  if (lane == 0) hn[rowid] = 0.5f * ss;
}

// ---------------------------------------------------------------------------
// One RVQ stage. One wave per block owning M=32 rows (two 16-row WMMA tiles)
// so every B fragment load feeds two v_wmma ops (2x arithmetic intensity on
// the codebook stream). Sweep all 1024 codes, running argmax, then gather.
// ---------------------------------------------------------------------------
__global__ __launch_bounds__(32)
void rvq_stage(const float* __restrict__ resid_in,
               const unsigned short* __restrict__ cbw,  // [L][NE][E] bf16 bits
               const float* __restrict__ hn,            // [L][NE]
               const float* __restrict__ cbf,           // [L][NE][E] f32
               float* __restrict__ xq_out,              // [B][E]
               float* __restrict__ idx_out,             // [B][L] (as float)
               float* __restrict__ resid_out,           // [B][E] stage slice
               float* __restrict__ partial,             // [gridDim] stage slice
               int stage) {
  __shared__ __align__(16) float raw[32][E_DIM];
  __shared__ int sidx[32];

  const int lane = threadIdx.x & 31;
  const int row  = lane & 15;     // M row within a 16-row tile (A layout)
  const int half = lane >> 4;     // K-half (A/B) / M-half (C)
  const int rowbase = blockIdx.x * 32;

  // ---- Phase 1: load raw residual rows to LDS, per-row 1/norm -------------
  float inv[2];
#pragma unroll
  for (int tt = 0; tt < 2; ++tt) {
    const int r32 = tt * 16 + row;
    const float* rsrc = resid_in + (size_t)(rowbase + r32) * E_DIM + half * 128;
    float ss = 0.f;
#pragma unroll
    for (int j = 0; j < 32; ++j) {
      float4 v = ((const float4*)rsrc)[j];
      ((float4*)&raw[r32][half * 128])[j] = v;
      ss += v.x * v.x + v.y * v.y + v.z * v.z + v.w * v.w;
    }
    ss += __shfl_xor(ss, 16, 32);                       // combine halves
    inv[tt] = 1.f / fmaxf(sqrtf(ss), 1e-12f);
  }
  __syncthreads();

  // ---- Phase 2: build bf16 A fragments for both 16x256 tiles --------------
  // A fragment as packed here (contiguous k, B matches; 16x32): lane<16: M=lane, V0-7 K=0..15;
  // lane>=16: same M, K offset +16.
  v16bf afrag[2][8];
#pragma unroll
  for (int tt = 0; tt < 2; ++tt) {
#pragma unroll
    for (int kk = 0; kk < 8; ++kk) {
      v16us au;
      const int kb = kk * 32 + half * 16;
      const int r32 = tt * 16 + row;
#pragma unroll
      for (int e = 0; e < 8; ++e) au[e]     = f2bf(raw[r32][kb + e] * inv[tt]);
#pragma unroll
      for (int e = 0; e < 8; ++e) au[8 + e] = f2bf(raw[r32][kb + 8 + e] * inv[tt]);
      afrag[tt][kk] = __builtin_bit_cast(v16bf, au);
    }
  }

  // ---- Phase 3: sweep all 1024 codes (16-code tiles); running argmax ------
  float best[2][8];
  int   bcode[2][8];
#pragma unroll
  for (int tt = 0; tt < 2; ++tt)
#pragma unroll
    for (int r = 0; r < 8; ++r) { best[tt][r] = -3.4e38f; bcode[tt][r] = 0; }

  const unsigned short* cbs = cbw + (size_t)stage * NE_CODES * E_DIM;
  const float*          hns = hn  + (size_t)stage * NE_CODES;

  for (int t = 0; t < 64; ++t) {
    const int code = t * 16 + row;                      // B layout: N = lane&15
    const unsigned short* brow = cbs + (size_t)code * E_DIM + half * 16;
    v8f acc0 = {0.f, 0.f, 0.f, 0.f, 0.f, 0.f, 0.f, 0.f};
    v8f acc1 = {0.f, 0.f, 0.f, 0.f, 0.f, 0.f, 0.f, 0.f};
#pragma unroll
    for (int kk = 0; kk < 8; ++kk) {
      v16bf bf = __builtin_bit_cast(v16bf, *(const v16us*)(brow + kk * 32));
      acc0 = __builtin_amdgcn_wmma_f32_16x16x32_bf16(false, afrag[0][kk], false, bf,
                                                     (short)0, acc0, false, false);
      acc1 = __builtin_amdgcn_wmma_f32_16x16x32_bf16(false, afrag[1][kk], false, bf,
                                                     (short)0, acc1, false, false);
    }
    const float h = hns[code];
#pragma unroll
    for (int r = 0; r < 8; ++r) {                       // C: lane holds N=lane&15, M=r+8*half
      const float s0 = acc0[r] - h;
      if (s0 > best[0][r]) { best[0][r] = s0; bcode[0][r] = code; }
      const float s1 = acc1[r] - h;
      if (s1 > best[1][r]) { best[1][r] = s1; bcode[1][r] = code; }
    }
  }
  // reduce argmax across the 16 code lanes sharing each row group
#pragma unroll
  for (int m = 1; m < 16; m <<= 1) {
#pragma unroll
    for (int tt = 0; tt < 2; ++tt)
#pragma unroll
      for (int r = 0; r < 8; ++r) {
        const float os = __shfl_xor(best[tt][r], m, 32);
        const int   oc = __shfl_xor(bcode[tt][r], m, 32);
        if (os > best[tt][r]) { best[tt][r] = os; bcode[tt][r] = oc; }
      }
  }
  if (row == 0) {                                       // lanes 0 and 16
#pragma unroll
    for (int tt = 0; tt < 2; ++tt)
#pragma unroll
      for (int r = 0; r < 8; ++r) {
        const int rr = tt * 16 + half * 8 + r;
        sidx[rr] = bcode[tt][r];
        idx_out[(size_t)(rowbase + rr) * L_STAGES + stage] = (float)bcode[tt][r];
      }
  }
  __syncthreads();

  // ---- Phase 4: gather f32 code row, loss partial, residual, x_q ----------
  float lsum = 0.f;
#pragma unroll
  for (int tt = 0; tt < 2; ++tt) {
    const int r32 = tt * 16 + row;
    const int myidx = sidx[r32];
    const float* crow = cbf + ((size_t)stage * NE_CODES + myidx) * E_DIM + half * 128;
    const size_t gofs = (size_t)(rowbase + r32) * E_DIM + half * 128;
    float*       ro   = resid_out + gofs;
    float*       xo   = xq_out + gofs;
    const float* rr   = &raw[r32][half * 128];
#pragma unroll 4
    for (int j = 0; j < 128; ++j) {
      const float c  = crow[j];
      const float rv = rr[j];
      const float rn = rv * inv[tt];
      const float d  = c - rn;
      lsum += d * d;
      ro[j] = rv - c;                                   // residual after stage
      if (stage == 0) xo[j] = c; else xo[j] += c;       // x_q accumulation
    }
  }
#pragma unroll
  for (int m = 16; m >= 1; m >>= 1) lsum += __shfl_xor(lsum, m, 32);
  if (lane == 0) partial[blockIdx.x] = lsum;
}

// ---------------------------------------------------------------------------
// Finalize: mean_loss = sum(partials) * (1+beta) / (L*B*E)
// ---------------------------------------------------------------------------
__global__ __launch_bounds__(256)
void rvq_finalize(const float* __restrict__ partial, int n,
                  float* __restrict__ out_loss) {
  __shared__ float sh[256];
  float s = 0.f;
  for (int i = threadIdx.x; i < n; i += 256) s += partial[i];
  sh[threadIdx.x] = s;
  __syncthreads();
  for (int st = 128; st > 0; st >>= 1) {
    if ((int)threadIdx.x < st) sh[threadIdx.x] += sh[threadIdx.x + st];
    __syncthreads();
  }
  if (threadIdx.x == 0)
    *out_loss = sh[0] * (1.f + BETA_F) /
                ((float)L_STAGES * (float)B_ROWS * (float)E_DIM);
}

// ---------------------------------------------------------------------------
extern "C" void kernel_launch(void* const* d_in, const int* in_sizes, int n_in,
                              void* d_out, int out_size, void* d_ws, size_t ws_size,
                              hipStream_t stream) {
  (void)in_sizes; (void)n_in; (void)out_size; (void)ws_size;
  const float* x  = (const float*)d_in[0];              // [B, E]
  const float* cb = (const float*)d_in[1];              // [L, NE, E]

  float* out       = (float*)d_out;
  float* xq_out    = out;                                   // [B*E]
  float* loss_out  = out + (size_t)B_ROWS * E_DIM;          // [1]
  float* idx_out   = loss_out + 1;                          // [B*L]
  float* resid_out = idx_out + (size_t)B_ROWS * L_STAGES;   // [L*B*E]

  unsigned short* cbw = (unsigned short*)d_ws;                              // 2 MB
  float* hn      = (float*)(cbw + (size_t)L_STAGES * NE_CODES * E_DIM);     // 16 KB
  float* partial = hn + (size_t)L_STAGES * NE_CODES;                        // 32 KB

  rvq_prep<<<(L_STAGES * NE_CODES) / 4, 128, 0, stream>>>(cb, cbw, hn);

  const int nblocks = B_ROWS / 32;                          // 2048 blocks, 1 wave each
  for (int l = 0; l < L_STAGES; ++l) {
    const float* rin = (l == 0) ? x : (resid_out + (size_t)(l - 1) * B_ROWS * E_DIM);
    rvq_stage<<<nblocks, 32, 0, stream>>>(rin, cbw, hn, cb,
        xq_out, idx_out,
        resid_out + (size_t)l * B_ROWS * E_DIM,
        partial + (size_t)l * nblocks, l);
  }
  rvq_finalize<<<1, 256, 0, stream>>>(partial, L_STAGES * nblocks, loss_out);
}
